// QLSTMPlus_65481071396715
// MI455X (gfx1250) — compile-verified, matrix operand order checked by reading, not run on hardware
//
#include <hip/hip_runtime.h>
#include <hip/hip_bf16.h>

typedef __attribute__((ext_vector_type(16))) _Float16 v16h;
typedef __attribute__((ext_vector_type(8)))  float    v8f;

// Problem dims (match reference)
#define TT   256
#define BSZ  128
#define IND  512
#define HID  512
#define COMB 1024
#define NG   2048   /* 4 gates * HID */

union FragH {
    v16h v;
    uint4 u[2];
};

// ---------------------------------------------------------------------------
// Prep kernels
// ---------------------------------------------------------------------------
__global__ void convert_f32_f16(const float* __restrict__ src,
                                _Float16* __restrict__ dst, int n) {
    int i = blockIdx.x * blockDim.x + threadIdx.x;
    int stride = gridDim.x * blockDim.x;
    for (; i < n; i += stride) dst[i] = (_Float16)src[i];
}

__global__ void pack_params(const float* __restrict__ Wf, const float* __restrict__ Wi,
                            const float* __restrict__ Wu, const float* __restrict__ Wo,
                            const float* __restrict__ bf_, const float* __restrict__ bi,
                            const float* __restrict__ bu, const float* __restrict__ bo,
                            const float* __restrict__ tf_, const float* __restrict__ ti,
                            const float* __restrict__ tu, const float* __restrict__ to_,
                            _Float16* __restrict__ Wcat,
                            float* __restrict__ bcat, float* __restrict__ tcat) {
    int tid = blockIdx.x * blockDim.x + threadIdx.x;
    int stride = gridDim.x * blockDim.x;
    const int WTOT = 4 * HID * COMB;  // 2M elements
    for (int i = tid; i < WTOT; i += stride) {
        int g = i >> 19;            // / (512*1024)
        int r = i & 0x7FFFF;        // % (512*1024)
        const float* src = (g == 0) ? Wf : (g == 1) ? Wi : (g == 2) ? Wu : Wo;
        Wcat[i] = (_Float16)src[r];
    }
    if (tid < NG) {
        int g = tid >> 9, j = tid & 511;
        const float* bsrc = (g == 0) ? bf_ : (g == 1) ? bi : (g == 2) ? bu : bo;
        const float* tsrc = (g == 0) ? tf_ : (g == 1) ? ti : (g == 2) ? tu : to_;
        bcat[tid] = bsrc[j];
        tcat[tid] = tsrc[j];
    }
}

__global__ void init_state(_Float16* __restrict__ H16, float* __restrict__ hF,
                           float* __restrict__ cF) {
    int i = blockIdx.x * blockDim.x + threadIdx.x;
    if (i < BSZ * HID) {
        H16[i] = (_Float16)0.0f;
        hF[i] = 0.0f;
        cF[i] = 0.0f;
    }
}

// ---------------------------------------------------------------------------
// Per-step gate GEMM:  Z[128,2048] = [x_t | h] (128x1024, f16) @ Wcat^T (f16)
// Wave computes a 16(M) x 64(N) tile: 1 A fragment reused by 4 B fragments
// per 32-wide K step -> 4x v_wmma_f32_16x16x32_f16 per A load.
// Grid: 32 blocks x 256 threads (8 waves); 256 wave-tiles total.
// ---------------------------------------------------------------------------
__global__ __launch_bounds__(256)
void gate_gemm(const _Float16* __restrict__ Xt,   // [128][512] f16 (timestep slice)
               const _Float16* __restrict__ H16,  // [128][512] f16 (h_{t-1})
               const _Float16* __restrict__ W,    // [2048][1024] f16
               float* __restrict__ Z)             // [128][2048] f32
{
    const int lane = threadIdx.x & 31;
    const int wave = threadIdx.x >> 5;
    const int tile = blockIdx.x * 8 + wave;   // 0..255
    const int mt = tile & 7;                  // 8 M-tiles of 16
    const int nt = tile >> 3;                 // 32 N-tiles of 64
    const int m0 = mt * 16;
    const int n0 = nt * 64;

    const int half  = lane >> 4;              // 0: K lo-chunk, 1: K hi-chunk
    const int row_a = m0 + (lane & 15);       // A row (M)
    const int row_b = n0 + (lane & 15);       // B "column" = row of W

    v8f acc0 = {}, acc1 = {}, acc2 = {}, acc3 = {};

    const _Float16* arow_x = Xt  + (size_t)row_a * IND;
    const _Float16* arow_h = H16 + (size_t)row_a * HID;
    const _Float16* brow   = W   + (size_t)row_b * COMB;

    for (int k0 = 0; k0 < COMB; k0 += 32) {
        // ---- A fragment: lane holds K = k0+half*8 .. +7 and +16 .. +23 ----
        const _Float16* ab = (k0 < IND) ? (arow_x + k0 + half * 8)
                                        : (arow_h + (k0 - IND) + half * 8);
        FragH a;
        a.u[0] = *(const uint4*)(ab);
        a.u[1] = *(const uint4*)(ab + 16);

        // ---- 4 B fragments (N tiles n0, n0+16, n0+32, n0+48) ----
        const _Float16* wb = brow + k0 + half * 8;
        FragH b0, b1, b2, b3;
        b0.u[0] = *(const uint4*)(wb);
        b0.u[1] = *(const uint4*)(wb + 16);
        b1.u[0] = *(const uint4*)(wb + 16 * COMB);
        b1.u[1] = *(const uint4*)(wb + 16 * COMB + 16);
        b2.u[0] = *(const uint4*)(wb + 32 * COMB);
        b2.u[1] = *(const uint4*)(wb + 32 * COMB + 16);
        b3.u[0] = *(const uint4*)(wb + 48 * COMB);
        b3.u[1] = *(const uint4*)(wb + 48 * COMB + 16);

        acc0 = __builtin_amdgcn_wmma_f32_16x16x32_f16(false, a.v, false, b0.v,
                                                      (short)0, acc0, false, false);
        acc1 = __builtin_amdgcn_wmma_f32_16x16x32_f16(false, a.v, false, b1.v,
                                                      (short)0, acc1, false, false);
        acc2 = __builtin_amdgcn_wmma_f32_16x16x32_f16(false, a.v, false, b2.v,
                                                      (short)0, acc2, false, false);
        acc3 = __builtin_amdgcn_wmma_f32_16x16x32_f16(false, a.v, false, b3.v,
                                                      (short)0, acc3, false, false);
    }

    // D layout: VGPR r -> lanes 0-15: M=m0+r, N=lane; lanes 16-31: M=m0+8+r, N=lane-16
    const int mbase = m0 + half * 8;
    const int ncol  = n0 + (lane & 15);
#pragma unroll
    for (int r = 0; r < 8; ++r) {
        float* p = Z + (size_t)(mbase + r) * NG + ncol;
        p[0]  = acc0[r];
        p[16] = acc1[r];
        p[32] = acc2[r];
        p[48] = acc3[r];
    }
}

// ---------------------------------------------------------------------------
// Per-step scan + LSTM update. One block (512 threads) per batch row.
// v = cos(z + b + theta) for 4 gates; inclusive product scan over HID=512
// (Hillis-Steele in LDS, 9 rounds); then gate activations + state update.
// ---------------------------------------------------------------------------
__device__ __forceinline__ float sigmoidf_(float x) {
    return 1.0f / (1.0f + __expf(-x));
}

__global__ __launch_bounds__(512)
void scan_update(const float* __restrict__ Z,     // [128][2048]
                 const float* __restrict__ bcat,  // [2048]
                 const float* __restrict__ tcat,  // [2048]
                 float* __restrict__ hF,          // [128][512]
                 float* __restrict__ cF,          // [128][512]
                 _Float16* __restrict__ H16,      // [128][512]
                 float* __restrict__ out_t)       // d_out + t*B*HID
{
    __shared__ float4 s[HID];
    const int b = blockIdx.x;
    const int j = threadIdx.x;
    const float* zr = Z + (size_t)b * NG;

    float4 v;
    v.x = __cosf(zr[j]          + bcat[j]          + tcat[j]);           // forget
    v.y = __cosf(zr[512 + j]    + bcat[512 + j]    + tcat[512 + j]);     // input
    v.z = __cosf(zr[1024 + j]   + bcat[1024 + j]   + tcat[1024 + j]);    // update
    v.w = __cosf(zr[1536 + j]   + bcat[1536 + j]   + tcat[1536 + j]);    // output
    s[j] = v;
    __syncthreads();

#pragma unroll
    for (int off = 1; off < HID; off <<= 1) {
        float4 prev = make_float4(1.0f, 1.0f, 1.0f, 1.0f);
        if (j >= off) prev = s[j - off];
        __syncthreads();
        v.x *= prev.x; v.y *= prev.y; v.z *= prev.z; v.w *= prev.w;
        s[j] = v;
        __syncthreads();
    }

    const float F = sigmoidf_(v.x);
    const float I = sigmoidf_(v.y);
    const float G = tanhf(v.z);
    const float O = sigmoidf_(v.w);

    const int idx = b * HID + j;
    const float c = F * cF[idx] + I * G;
    const float h = O * tanhf(c);
    cF[idx] = c;
    hF[idx] = h;
    H16[idx] = (_Float16)h;
    out_t[idx] = h;
}

__global__ void final_copy(const float* __restrict__ hF, const float* __restrict__ cF,
                           float* __restrict__ dst) {
    int i = blockIdx.x * blockDim.x + threadIdx.x;
    if (i < BSZ * HID) {
        dst[i] = hF[i];
        dst[BSZ * HID + i] = cF[i];
    }
}

// ---------------------------------------------------------------------------
// Launch
// ---------------------------------------------------------------------------
extern "C" void kernel_launch(void* const* d_in, const int* in_sizes, int n_in,
                              void* d_out, int out_size, void* d_ws, size_t ws_size,
                              hipStream_t stream) {
    const float* x  = (const float*)d_in[0];
    const float* Wf = (const float*)d_in[1];
    const float* bf_ = (const float*)d_in[2];
    const float* tf_ = (const float*)d_in[3];
    const float* Wi = (const float*)d_in[4];
    const float* bi = (const float*)d_in[5];
    const float* ti = (const float*)d_in[6];
    const float* Wu = (const float*)d_in[7];
    const float* bu = (const float*)d_in[8];
    const float* tu = (const float*)d_in[9];
    const float* Wo = (const float*)d_in[10];
    const float* bo = (const float*)d_in[11];
    const float* to_ = (const float*)d_in[12];
    float* out = (float*)d_out;

    // Workspace layout (bytes)
    char* ws = (char*)d_ws;
    _Float16* X16  = (_Float16*)(ws);                          // 256*128*512*2 = 33,554,432
    _Float16* Wcat = (_Float16*)(ws + 33554432);               // 2048*1024*2   =  4,194,304
    float*    bcat = (float*)   (ws + 37748736);               // 2048*4
    float*    tcat = (float*)   (ws + 37756928);               // 2048*4
    _Float16* H16  = (_Float16*)(ws + 37765120);               // 128*512*2
    float*    cF   = (float*)   (ws + 37896192);               // 128*512*4
    float*    Zbuf = (float*)   (ws + 38158336);               // 128*2048*4 = 1,048,576
    float*    hF   = (float*)   (ws + 39206912);               // 128*512*4

    // Prep: input/weight conversion, state init
    convert_f32_f16<<<4096, 256, 0, stream>>>(x, X16, TT * BSZ * IND);
    pack_params<<<2048, 256, 0, stream>>>(Wf, Wi, Wu, Wo, bf_, bi, bu, bo,
                                          tf_, ti, tu, to_, Wcat, bcat, tcat);
    init_state<<<(BSZ * HID + 255) / 256, 256, 0, stream>>>(H16, hF, cF);

    // Recurrent loop: GEMM + scan/update per timestep (stream-ordered chain)
    for (int t = 0; t < TT; ++t) {
        gate_gemm<<<32, 256, 0, stream>>>(X16 + (size_t)t * BSZ * IND, H16, Wcat, Zbuf);
        scan_update<<<BSZ, HID, 0, stream>>>(Zbuf, bcat, tcat, hF, cF, H16,
                                             out + (size_t)t * BSZ * HID);
    }

    // (hx, cx) tail
    final_copy<<<(BSZ * HID + 255) / 256, 256, 0, stream>>>(hF, cF,
                                                           out + (size_t)TT * BSZ * HID);
}
